// VQVAE_31911607009332
// MI455X (gfx1250) — matrix operand order repaired and checked by reading, not compiled, not run on hardware
//
#include <hip/hip_runtime.h>
#include <hip/hip_bf16.h>
#include <math.h>

// ---------------------------------------------------------------------------
// VQ-VAE forward for MI455X (gfx1250, wave32).
//  - Big GEMMs: v_wmma_f32_16x16x32_bf16 (f32 accumulate), double-buffered LDS.
//  - Small / precision-sensitive GEMMs: v_wmma_f32_16x16x4_f32 with
//    global_load_async_to_lds_b128 staging (ASYNCcnt pipelined).
//  - Cosine argmax: only codebook rows normalized (query row scaling does not
//    change argmax), wave-per-row shuffle reduction.
// ---------------------------------------------------------------------------

typedef __attribute__((ext_vector_type(16))) __bf16 v16bf;
typedef __attribute__((ext_vector_type(8)))  float  v8f;
typedef __attribute__((ext_vector_type(2)))  float  v2f;

union FragBF { uint4 q[2]; v16bf v; };

// ===========================================================================
// bf16 WMMA GEMM + bias + relu.  C[M,N] = relu(A[M,K] @ W[K,N] + bias)
// 256 threads (8 waves), workgroup tile 128x128, wave tile 16x128, K step 32.
// Double-buffered LDS (global->regs for tile k+1 issued before compute of k).
// A row-major in LDS, W transposed in LDS so each lane's v16bf fragment is
// two 16B ds loads (contiguous k = 16h + s order, same in A and B).
// ===========================================================================
__global__ __launch_bounds__(256)
void gemm_bf16_relu(const float* __restrict__ A, const float* __restrict__ W,
                    const float* __restrict__ bias, float* __restrict__ C,
                    int M, int K, int N, int nt) {
  __shared__ __bf16 lA[2][128][40];   // [m][k], 80B row stride (16B aligned)
  __shared__ __bf16 lB[2][128][40];   // transposed: [n][k]

  const int tid  = threadIdx.x;
  const int wave = tid >> 5, lane = tid & 31;
  const int laneM = lane & 15, kh = lane >> 4;
  const int m0 = blockIdx.y * 128, n0 = blockIdx.x * 128;

  v8f acc[8] = {};

  // staging: A 128x32 -> thread = (row, 16-col half); W 32x128 -> thread =
  // (k-pair, 8-col group); 16 consecutive lanes read 512B of one W row.
  const int ar  = tid >> 1, ac  = (tid & 1) * 16;
  const int bkk = tid >> 4, bn8 = (tid & 15) * 8;

  const float* aptr = A + (size_t)(m0 + ar) * K + ac;
  const float* bptr = W + (size_t)(2 * bkk) * N + n0 + bn8;

  float4 areg[4], breg[4];   // breg[0..1] = row 2k, breg[2..3] = row 2k+1

  auto gload = [&](int k0) {
    const float4* ap = (const float4*)(aptr + k0);
    #pragma unroll
    for (int j = 0; j < 4; ++j) areg[j] = ap[j];
    const float4* bp0 = (const float4*)(bptr + (size_t)k0 * N);
    const float4* bp1 = (const float4*)(bptr + (size_t)(k0 + 1) * N);
    breg[0] = bp0[0]; breg[1] = bp0[1];
    breg[2] = bp1[0]; breg[3] = bp1[1];
  };

  auto lstore = [&](int buf) {
    union { __bf16 h[16]; uint4 q[2]; } pa;
    #pragma unroll
    for (int j = 0; j < 4; ++j) {
      pa.h[4 * j + 0] = (__bf16)areg[j].x; pa.h[4 * j + 1] = (__bf16)areg[j].y;
      pa.h[4 * j + 2] = (__bf16)areg[j].z; pa.h[4 * j + 3] = (__bf16)areg[j].w;
    }
    *(uint4*)&lA[buf][ar][ac]     = pa.q[0];
    *(uint4*)&lA[buf][ar][ac + 8] = pa.q[1];
    const float* r0 = (const float*)&breg[0];
    const float* r1 = (const float*)&breg[2];
    #pragma unroll
    for (int j = 0; j < 8; ++j) {       // pack (k, k+1) pair -> one b32 store
      union { __bf16 h[2]; unsigned u; } p;
      p.h[0] = (__bf16)r0[j]; p.h[1] = (__bf16)r1[j];
      *(unsigned*)&lB[buf][bn8 + j][2 * bkk] = p.u;
    }
  };

  const int mrow = wave * 16 + laneM;

  auto compute = [&](int buf) {
    FragBF fa;      // lanes 0-15: K {0..15}; lanes 16-31: K {16..31}
    fa.q[0] = *(const uint4*)&lA[buf][mrow][16 * kh];
    fa.q[1] = *(const uint4*)&lA[buf][mrow][16 * kh + 8];
    FragBF fb[8];   // lane n holds K {0..15} (+16 for hi lanes)
    #pragma unroll
    for (int nb = 0; nb < 8; ++nb) {
      const int nrow = nb * 16 + laneM;
      fb[nb].q[0] = *(const uint4*)&lB[buf][nrow][16 * kh];
      fb[nb].q[1] = *(const uint4*)&lB[buf][nrow][16 * kh + 8];
    }
    #pragma unroll
    for (int nb = 0; nb < 8; ++nb)
      acc[nb] = __builtin_amdgcn_wmma_f32_16x16x32_bf16(
          false, fa.v, false, fb[nb].v, (short)0, acc[nb], false, false);
  };

  const int nK = K >> 5;
  gload(0); lstore(0);
  __syncthreads();
  for (int kt = 0; kt < nK; ++kt) {
    const int cur = kt & 1;
    if (kt + 1 < nK) { gload((kt + 1) << 5); lstore(cur ^ 1); }
    compute(cur);
    __syncthreads();
  }

  #pragma unroll
  for (int nb = 0; nb < 8; ++nb) {
    const int col = n0 + nb * 16 + laneM;
    const float bv = bias[col];
    #pragma unroll
    for (int v = 0; v < 8; ++v) {
      const int row = m0 + wave * 16 + v + 8 * kh;   // C layout: M = v (+8 hi)
      float c = acc[nb][v] + bv;
      c = c > 0.0f ? c : 0.0f;
      float* p = C + (size_t)row * N + col;
      if (nt) __builtin_nontemporal_store(c, p); else *p = c;
    }
  }
}

// ===========================================================================
// f32 WMMA GEMM (16x16x4), optional bias / relu. Workgroup tile 128x64,
// wave tile 16x64, K step 16. Staging via GLOBAL_LOAD_ASYNC_TO_LDS_B128
// (ASYNCcnt), double-buffered: tile k+1 in flight during compute of tile k.
// ===========================================================================
__global__ __launch_bounds__(256)
void gemm_f32(const float* __restrict__ A, const float* __restrict__ B,
              const float* __restrict__ bias, float* __restrict__ C,
              int M, int K, int N, int relu) {
  __shared__ float lA[2][128][20];   // [m][k], 80B row stride
  __shared__ float lB[2][16][68];    // [k][n], 272B row stride

  const int tid  = threadIdx.x;
  const int wave = tid >> 5, lane = tid & 31;
  const int laneM = lane & 15, kh = lane >> 4;
  const int m0 = blockIdx.y * 128, n0 = blockIdx.x * 64;

  v8f acc[4] = {};

  const int ar = tid >> 1, ac = (tid & 1) * 8;     // A: 128 x 16
  const int bk = tid >> 4, bn = (tid & 15) * 4;    // B: 16 x 64

  const float* aptr = A + (size_t)(m0 + ar) * K + ac;
  const float* bptr = B + (size_t)bk * N + n0 + bn;

  typedef __attribute__((address_space(3))) char* as3ptr;

  auto stage_async = [&](int k0, int buf) {
    const float* ga = aptr + k0;                      // 8 floats (2 x b128)
    unsigned la = (unsigned)(size_t)(as3ptr)(void*)&lA[buf][ar][ac];
    asm volatile("global_load_async_to_lds_b128 %0, %1, off"
                 :: "v"(la), "v"(ga) : "memory");
    asm volatile("global_load_async_to_lds_b128 %0, %1, off offset:16"
                 :: "v"(la), "v"(ga) : "memory");
    const float* gb = bptr + (size_t)k0 * N;          // 4 floats (1 x b128)
    unsigned lb = (unsigned)(size_t)(as3ptr)(void*)&lB[buf][bk][bn];
    asm volatile("global_load_async_to_lds_b128 %0, %1, off"
                 :: "v"(lb), "v"(gb) : "memory");
  };

  const int mrow = wave * 16 + laneM;

  auto compute = [&](int buf) {
    #pragma unroll
    for (int kk = 0; kk < 16; kk += 4) {
      v2f a = *(const v2f*)&lA[buf][mrow][kk + 2 * kh];   // K = kk+2*kh, +1
      v2f b[4];
      #pragma unroll
      for (int nb = 0; nb < 4; ++nb) {
        b[nb].x = lB[buf][kk + 2 * kh][nb * 16 + laneM];
        b[nb].y = lB[buf][kk + 2 * kh + 1][nb * 16 + laneM];
      }
      #pragma unroll
      for (int nb = 0; nb < 4; ++nb)
        acc[nb] = __builtin_amdgcn_wmma_f32_16x16x4_f32(
            false, a, false, b[nb], (short)0, acc[nb], false, false);
    }
  };

  const int nK = K >> 4;
  stage_async(0, 0);
  asm volatile("s_wait_asynccnt 0x0" ::: "memory");
  __syncthreads();
  for (int kt = 0; kt < nK; ++kt) {
    const int cur = kt & 1;
    const bool more = (kt + 1 < nK);
    if (more) stage_async((kt + 1) << 4, cur ^ 1);   // in flight during compute
    compute(cur);
    if (more) asm volatile("s_wait_asynccnt 0x0" ::: "memory");
    __syncthreads();
  }

  #pragma unroll
  for (int nb = 0; nb < 4; ++nb) {
    const int col = n0 + nb * 16 + laneM;
    const float bv = bias ? bias[col] : 0.0f;
    #pragma unroll
    for (int v = 0; v < 8; ++v) {
      const int row = m0 + wave * 16 + v + 8 * kh;
      float c = acc[nb][v] + bv;
      if (relu) c = c > 0.0f ? c : 0.0f;
      C[(size_t)row * N + col] = c;
    }
  }
}

// ===========================================================================
// Codebook prep: copy emb -> output, write row-normalized transpose [256][512]
// so sim = encoded @ embnT. One wave per codebook row.
// ===========================================================================
__global__ __launch_bounds__(256)
void emb_prep(const float* __restrict__ emb, float* __restrict__ emb_out,
              float* __restrict__ embnT) {
  const int wave = threadIdx.x >> 5, lane = threadIdx.x & 31;
  const int row = blockIdx.x * 8 + wave;            // < 512
  float v[8];
  float ss = 0.0f;
  #pragma unroll
  for (int i = 0; i < 8; ++i) {
    const int c = lane + i * 32;
    v[i] = emb[row * 256 + c];
    emb_out[row * 256 + c] = v[i];
    ss += v[i] * v[i];
  }
  #pragma unroll
  for (int off = 16; off > 0; off >>= 1) ss += __shfl_xor(ss, off, 32);
  const float sc = 1.0f / (sqrtf(ss) + 1e-12f);
  #pragma unroll
  for (int i = 0; i < 8; ++i) embnT[(size_t)(lane + i * 32) * 512 + row] = v[i] * sc;
}

// ===========================================================================
// Wave-per-row argmax over sim[8192,512] -> one-hot + codebook gather.
// Tie-break matches jnp.argmax (first/lowest index).
// ===========================================================================
__global__ __launch_bounds__(256)
void vq_select(const float* __restrict__ sim, const float* __restrict__ emb,
               float* __restrict__ vq, float* __restrict__ onehot) {
  const int wave = threadIdx.x >> 5, lane = threadIdx.x & 31;
  const int row = blockIdx.x * 8 + wave;            // < 8192
  const float* s = sim + (size_t)row * 512;
  float best = -3.402823466e38f;
  int bi = 0x7fffffff;
  for (int i = lane; i < 512; i += 32) {
    const float x = s[i];
    if (x > best) { best = x; bi = i; }
  }
  #pragma unroll
  for (int off = 16; off > 0; off >>= 1) {
    const float ob = __shfl_xor(best, off, 32);
    const int   oi = __shfl_xor(bi, off, 32);
    if (ob > best || (ob == best && oi < bi)) { best = ob; bi = oi; }
  }
  for (int i = lane; i < 512; i += 32)
    onehot[(size_t)row * 512 + i] = (i == bi) ? 1.0f : 0.0f;
  #pragma unroll
  for (int j = 0; j < 8; ++j) {
    const int c = lane + j * 32;
    vq[(size_t)row * 256 + c] = emb[(size_t)bi * 256 + c];
  }
}

// ===========================================================================
extern "C" void kernel_launch(void* const* d_in, const int* in_sizes, int n_in,
                              void* d_out, int out_size, void* d_ws, size_t ws_size,
                              hipStream_t stream) {
  const float* x      = (const float*)d_in[0];
  const float* enc_w1 = (const float*)d_in[1];
  const float* enc_b1 = (const float*)d_in[2];
  const float* enc_w2 = (const float*)d_in[3];
  const float* enc_b2 = (const float*)d_in[4];
  const float* emb    = (const float*)d_in[5];
  const float* dec_w1 = (const float*)d_in[6];
  const float* dec_b1 = (const float*)d_in[7];
  const float* dec_w2 = (const float*)d_in[8];
  const float* dec_b2 = (const float*)d_in[9];

  constexpr int B = 8192, FEAT = 4096, MID = 512, EMB = 256, KCODE = 512;

  float* out     = (float*)d_out;
  float* encoded = out;                          // [8192,256]
  float* vqf     = out + 2097152;                // [8192,256]
  float* onehot  = out + 4194304;                // [8192,512]
  float* decoded = out + 8388608;                // [8192,4096]
  float* emb_out = out + 41943040;               // [512,256]

  float* wsf   = (float*)d_ws;
  float* buf0  = wsf;              // reused: h -> sim -> decoder hidden (16MB)
  float* embnT = wsf + 4194304;    // [256][512] normalized codebook^T

  // 1) codebook prep (copies emb to output too)
  emb_prep<<<dim3(KCODE / 8), dim3(256), 0, stream>>>(emb, emb_out, embnT);
  // 2) h = relu(x @ enc_w1 + b1)          [bf16 WMMA, 34.4 GFLOP]
  gemm_bf16_relu<<<dim3(MID / 128, B / 128), dim3(256), 0, stream>>>(
      x, enc_w1, enc_b1, buf0, B, FEAT, MID, 0);
  // 3) encoded = h @ enc_w2 + b2          [f32 WMMA, async staging]
  gemm_f32<<<dim3(EMB / 64, B / 128), dim3(256), 0, stream>>>(
      buf0, enc_w2, enc_b2, encoded, B, MID, EMB, 0);
  // 4) sim = encoded @ embnT              [f32 WMMA, argmax-exact]
  gemm_f32<<<dim3(KCODE / 64, B / 128), dim3(256), 0, stream>>>(
      encoded, embnT, nullptr, buf0, B, EMB, KCODE, 0);
  // 5) argmax + one-hot + codebook gather
  vq_select<<<dim3(B / 8), dim3(256), 0, stream>>>(buf0, emb, vqf, onehot);
  // 6) d = relu(vq_feat @ dec_w1 + b1)    [f32 WMMA]
  gemm_f32<<<dim3(MID / 64, B / 128), dim3(256), 0, stream>>>(
      vqf, dec_w1, dec_b1, buf0, B, EMB, MID, 1);
  // 7) decoded = relu(d @ dec_w2 + b2)    [bf16 WMMA, NT stores, 34.4 GFLOP]
  gemm_bf16_relu<<<dim3(FEAT / 128, B / 128), dim3(256), 0, stream>>>(
      buf0, dec_w2, dec_b2, decoded, B, MID, FEAT, 1);
}
